// CrossAttFA_73375221285086
// MI455X (gfx1250) — matrix operand order repaired and checked by reading, not compiled, not run on hardware
//
#include <hip/hip_runtime.h>
#include <hip/hip_bf16.h>
#include <stdint.h>

typedef _Float16 h16v __attribute__((ext_vector_type(16)));
typedef _Float16 h8v  __attribute__((ext_vector_type(8)));
typedef float    f8v  __attribute__((ext_vector_type(8)));

#define BATCH 2
#define CIN   64
#define ANUM  25
#define HW    48
#define PIX   (HW*HW)        // 2304
#define OC    8
#define LROW  PIX            // 2304 rows per batch
#define DRAW  1800           // 25 * 8 * 9
#define DPAD  1824           // padded to multiple of 32 (zero-filled)
#define KSTEPS (DPAD/32)     // 57

// ---------------------------------------------------------------------------
// 1) 1x1 conv: q[n,o,pix] = sum_c W[o,c] * x[b,c,a,pix]   (n = b*25+a)
// ---------------------------------------------------------------------------
__global__ void ca_conv1x1(const float* __restrict__ x, const float* __restrict__ W,
                           float* __restrict__ out) {
    int t = blockIdx.x * blockDim.x + threadIdx.x;
    const int total = BATCH * ANUM * OC * PIX;
    if (t >= total) return;
    int pix = t % PIX;
    int o   = (t / PIX) % OC;
    int n   = t / (PIX * OC);
    int b = n / ANUM, a = n % ANUM;
    float acc = 0.f;
#pragma unroll 8
    for (int c = 0; c < CIN; ++c)
        acc = fmaf(W[o * CIN + c],
                   x[(((size_t)(b * CIN + c)) * ANUM + a) * PIX + pix], acc);
    out[((size_t)n * OC + o) * PIX + pix] = acc;
}

// ---------------------------------------------------------------------------
// 2) Build one L2-normalized f16 row of the Q/K matrix per workgroup.
//    Row layout d = a*72 + o*9 + (i*3+j); elements [1800,1824) are zero pad.
// ---------------------------------------------------------------------------
__global__ void ca_build_rows(const float* __restrict__ src, _Float16* __restrict__ mat) {
    __shared__ float vals[DPAD];
    __shared__ float red[8];
    int row = blockIdx.x;            // b*LROW + n
    int b = row / LROW, n = row % LROW;
    int y = n / HW, x = n % HW;
    int tid = threadIdx.x;

    float ss = 0.f;
    for (int d = tid; d < DPAD; d += blockDim.x) {
        float v = 0.f;
        if (d < DRAW) {
            int a  = d / (OC * 9);
            int r  = d % (OC * 9);
            int o  = r / 9, ij = r % 9;
            int i = ij / 3, j = ij % 3;
            int yy = y + i - 1, xx = x + j - 1;
            if (yy >= 0 && yy < HW && xx >= 0 && xx < HW)
                v = src[((size_t)((b * ANUM + a) * OC) + o) * PIX + yy * HW + xx];
        }
        vals[d] = v;
        ss += v * v;
    }
    // wave32 reduce, then cross-wave via LDS
    for (int m = 1; m < 32; m <<= 1) ss += __shfl_xor(ss, m, 32);
    if ((tid & 31) == 0) red[tid >> 5] = ss;
    __syncthreads();
    if (tid == 0) {
        float tot = 0.f;
        for (int w = 0; w < 8; ++w) tot += red[w];
        red[0] = tot;
    }
    __syncthreads();
    float scale = 1.f / fmaxf(sqrtf(red[0]), 1e-12f);
    for (int d = tid; d < DPAD; d += blockDim.x)
        mat[(size_t)row * DPAD + d] = (_Float16)(vals[d] * scale);
}

// ---------------------------------------------------------------------------
// 3) attn = Q*K^T with fused per-row argmax, f16 WMMA 16x16x32.
//    Each wave owns two 16-row M tiles (B fragment shared), scans all N tiles.
// ---------------------------------------------------------------------------
__global__ void __launch_bounds__(256)
ca_attn_argmax(const _Float16* __restrict__ Qm, const _Float16* __restrict__ Km,
               int* __restrict__ idx_out) {
    const int lane   = threadIdx.x & 31;
    const int halfid = lane >> 4;     // 0: lanes 0-15, 1: lanes 16-31
    const int l15    = lane & 15;

    int g = blockIdx.x * (blockDim.x >> 5) + (threadIdx.x >> 5);
    const int pairsPerB = (LROW / 16) / 2;        // 72
    int b  = g / pairsPerB;
    int pr = g % pairsPerB;
    if (b >= BATCH) return;

    const _Float16* Qb = Qm + (size_t)b * LROW * DPAD;
    const _Float16* Kb = Km + (size_t)b * LROW * DPAD;
    const int mb0 = pr * 32;
    const int mb1 = mb0 + 16;

    float bestv[2][8];
    int   besti[2][8];
#pragma unroll
    for (int t = 0; t < 2; ++t)
#pragma unroll
        for (int r = 0; r < 8; ++r) { bestv[t][r] = -3.4e38f; besti[t][r] = 0; }

    // A fragment lane base: row = mbase + l15, chunks at K {0..7,16..23}+8*half
    const _Float16* Arow0 = Qb + (size_t)(mb0 + l15) * DPAD + (halfid ? 8 : 0);
    const _Float16* Arow1 = Qb + (size_t)(mb1 + l15) * DPAD + (halfid ? 8 : 0);

    for (int nt = 0; nt < LROW / 16; ++nt) {
        const int nbase = nt * 16;
        // B fragment lane base: col = K-row (nbase+l15), chunks at K {0..7,16..23}+8*half
        const _Float16* Brow = Kb + (size_t)(nbase + l15) * DPAD + (halfid ? 8 : 0);

        f8v acc0 = {}; f8v acc1 = {};
        for (int kt = 0; kt < KSTEPS; ++kt) {
            const int k0 = kt * 32;
            h8v blo = *(const h8v*)(Brow + k0);
            h8v bhi = *(const h8v*)(Brow + k0 + 16);
            h16v bf;
#pragma unroll
            for (int e = 0; e < 8; ++e) { bf[e] = blo[e]; bf[8 + e] = bhi[e]; }

            h8v a0lo = *(const h8v*)(Arow0 + k0);
            h8v a0hi = *(const h8v*)(Arow0 + k0 + 16);
            h16v af0;
#pragma unroll
            for (int e = 0; e < 8; ++e) { af0[e] = a0lo[e]; af0[8 + e] = a0hi[e]; }
            acc0 = __builtin_amdgcn_wmma_f32_16x16x32_f16(
                false, af0, false, bf, (short)0, acc0, false, false);

            h8v a1lo = *(const h8v*)(Arow1 + k0);
            h8v a1hi = *(const h8v*)(Arow1 + k0 + 16);
            h16v af1;
#pragma unroll
            for (int e = 0; e < 8; ++e) { af1[e] = a1lo[e]; af1[8 + e] = a1hi[e]; }
            acc1 = __builtin_amdgcn_wmma_f32_16x16x32_f16(
                false, af1, false, bf, (short)0, acc1, false, false);
        }

        // Fused argmax epilogue: slot r = (row r+8*half, col nbase+l15).
        const int nglob = nbase + l15;
#pragma unroll
        for (int r = 0; r < 8; ++r) {
            float v0 = acc0[r];
            if (v0 > bestv[0][r]) { bestv[0][r] = v0; besti[0][r] = nglob; }
            float v1 = acc1[r];
            if (v1 > bestv[1][r]) { bestv[1][r] = v1; besti[1][r] = nglob; }
        }
    }

    // Cross-lane argmax over the 16 columns of each half (first-occurrence ties).
#pragma unroll
    for (int t = 0; t < 2; ++t) {
#pragma unroll
        for (int r = 0; r < 8; ++r) {
            float v = bestv[t][r];
            int   i = besti[t][r];
            for (int m = 1; m < 16; m <<= 1) {
                float ov = __shfl_xor(v, m, 32);
                int   oi = __shfl_xor(i, m, 32);
                if (ov > v || (ov == v && oi < i)) { v = ov; i = oi; }
            }
            if (l15 == 0) {
                int mrow = (t == 0 ? mb0 : mb1) + r + 8 * halfid;
                idx_out[(size_t)b * LROW + mrow] = i;
            }
        }
    }
}

// ---------------------------------------------------------------------------
// 4) Fused V-gather + fold: out[b,c,a,y,x] = sum over 3x3 offsets of
//    x3[b,c,a, pos(idx[neighbor]) + offset]  (V never materialized).
// ---------------------------------------------------------------------------
__global__ void ca_gather_fold(const float* __restrict__ x3, const int* __restrict__ idxp,
                               float* __restrict__ out) {
    int t = blockIdx.x * blockDim.x + threadIdx.x;
    const int total = BATCH * CIN * ANUM * PIX;
    if (t >= total) return;
    int x = t % HW;
    int y = (t / HW) % HW;
    int a = (t / PIX) % ANUM;
    int c = (t / (PIX * ANUM)) % CIN;
    int b = t / (PIX * ANUM * CIN);

    const float* vimg = x3 + (((size_t)(b * CIN + c)) * ANUM + a) * PIX;
    const int* idxb = idxp + (size_t)b * LROW;

    float sum = 0.f;
#pragma unroll
    for (int i = 0; i < 3; ++i)
#pragma unroll
        for (int j = 0; j < 3; ++j) {
            int yo = y - i + 1, xo = x - j + 1;
            if (yo < 0 || yo >= HW || xo < 0 || xo >= HW) continue;
            int m  = idxb[yo * HW + xo];
            int ym = m / HW, xm = m % HW;
            int yy = ym + i - 1, xx = xm + j - 1;
            if (yy >= 0 && yy < HW && xx >= 0 && xx < HW)
                sum += vimg[yy * HW + xx];
        }
    out[t] = sum;   // t enumerates (b,c,a,y,x) = output flat order
}

// ---------------------------------------------------------------------------
extern "C" void kernel_launch(void* const* d_in, const int* in_sizes, int n_in,
                              void* d_out, int out_size, void* d_ws, size_t ws_size,
                              hipStream_t stream) {
    const float* x1 = (const float*)d_in[0];
    const float* x2 = (const float*)d_in[1];
    const float* x3 = (const float*)d_in[2];
    const float* W  = (const float*)d_in[3];

    char* ws = (char*)d_ws;
    const size_t SZ_QK   = (size_t)BATCH * ANUM * OC * PIX * sizeof(float);   // 3.69 MB
    const size_t SZ_MAT  = (size_t)BATCH * LROW * DPAD * 2u;                  // 16.8 MB (f16)
    const size_t QBUF_OFF = 0;
    const size_t KBUF_OFF = QBUF_OFF + SZ_QK;
    const size_t QMAT_OFF = KBUF_OFF + SZ_QK;
    const size_t KMAT_OFF = QMAT_OFF + SZ_MAT;
    const size_t IDX_OFF  = KMAT_OFF + SZ_MAT;

    float*     qbuf = (float*)(ws + QBUF_OFF);
    float*     kbuf = (float*)(ws + KBUF_OFF);
    _Float16*  qmat = (_Float16*)(ws + QMAT_OFF);
    _Float16*  kmat = (_Float16*)(ws + KMAT_OFF);
    int*       idxp = (int*)(ws + IDX_OFF);

    const int convTotal = BATCH * ANUM * OC * PIX;            // 921600
    ca_conv1x1<<<(convTotal + 255) / 256, 256, 0, stream>>>(x1, W, qbuf);
    ca_conv1x1<<<(convTotal + 255) / 256, 256, 0, stream>>>(x2, W, kbuf);

    ca_build_rows<<<BATCH * LROW, 256, 0, stream>>>(qbuf, qmat);
    ca_build_rows<<<BATCH * LROW, 256, 0, stream>>>(kbuf, kmat);

    const int totalWaves = BATCH * (LROW / 16) / 2;           // 144 waves
    ca_attn_argmax<<<totalWaves / 8, 256, 0, stream>>>(qmat, kmat, idxp);

    const int foldTotal = BATCH * CIN * ANUM * PIX;           // 7,372,800
    ca_gather_fold<<<(foldTotal + 255) / 256, 256, 0, stream>>>(x3, idxp, (float*)d_out);
}
